// CVMultiHeadAttention_81544249081936
// MI455X (gfx1250) — matrix operand order repaired and checked by reading, not compiled, not run on hardware
//
#include <hip/hip_runtime.h>

typedef __attribute__((ext_vector_type(16))) __bf16          v16bf;
typedef __attribute__((ext_vector_type(8)))  __bf16          v8bf;
typedef __attribute__((ext_vector_type(16))) unsigned short  v16u;
typedef __attribute__((ext_vector_type(8)))  float           v8f;

constexpr int B_ = 4, S_ = 1024, D_ = 128, H_ = 8;
constexpr int HD    = H_ * D_;        // 1024
constexpr int MROWS = B_ * S_;        // 4096

// ---------- helpers ----------

__device__ inline __bf16 f2bf(float f) {
  unsigned u = __builtin_bit_cast(unsigned, f);
  u += 0x7FFFu + ((u >> 16) & 1u);                 // round-to-nearest-even
  unsigned short h = (unsigned short)(u >> 16);
  return __builtin_bit_cast(__bf16, h);
}

__device__ inline v16bf negbf(v16bf x) {
  v16u u = __builtin_bit_cast(v16u, x);
  u ^= (unsigned short)0x8000;
  return __builtin_bit_cast(v16bf, u);
}

// A fragment: 16x32 bf16 tile (MxK), row-major source with leading dim ld.
// Lane l: row = l&15; lanes 0-15 hold K 0..15, lanes 16-31 K 16..31.
template <typename PT>
__device__ inline v16bf load_a(PT base, int ld, int lane) {
  int row = lane & 15;
  int ko  = (lane < 16) ? 0 : 16;
  return *(const v16bf*)(base + row * ld + ko);
}

// B fragment: 32x16 (KxN) where B(k,n) = src[n*ld + k] (each of the 16
// columns is a contiguous row of src). Lanes 0-15: K 0..15; lanes 16-31: K 16..31.
__device__ inline v16bf load_b(const __bf16* base, int ld, int lane) {
  int col = lane & 15;
  int ko  = (lane < 16) ? 0 : 16;
  return *(const v16bf*)(base + col * ld + ko);
}

#define WMMA_BF16(A, B, C) \
  __builtin_amdgcn_wmma_f32_16x16x32_bf16(false, (A), false, (B), (short)0, (C), false, false)

// ---------- fp32 -> bf16 conversion ----------

__global__ void cvt_kernel(const float* __restrict__ src, __bf16* __restrict__ dst, int n) {
  int i = blockIdx.x * blockDim.x + threadIdx.x;
  if (i < n) dst[i] = f2bf(src[i]);
}

// ---------- complex GEMM: Y = (Xr + iXi) @ (Wr + iWi)^T ----------
// X: [M,K] row-major bf16, W: [N,K] row-major bf16 (torch Linear convention).
// mode 0: write bf16 outputs head-major [B,H,S,D]      (Q/K projections)
// mode 1: write bf16 outputs transposed [B,H,D,S]      (V projection)
// mode 2: write interleaved complex64 floats [M,N,2]   (output projection)

__global__ __launch_bounds__(128) void cgemm_kernel(
    const __bf16* __restrict__ Ar, const __bf16* __restrict__ Ai,
    const __bf16* __restrict__ Wr, const __bf16* __restrict__ Wi,
    int K, int mode,
    __bf16* __restrict__ outr, __bf16* __restrict__ outi,
    float* __restrict__ outc)
{
  const int lane = threadIdx.x & 31;
  const int wave = threadIdx.x >> 5;
  const int m0 = blockIdx.x * 64 + wave * 16;
  const int n0 = blockIdx.y * 64;

  v8f accA[4], accB[4], accI[4];
  const v8f vz = {};
#pragma unroll
  for (int t = 0; t < 4; ++t) { accA[t] = vz; accB[t] = vz; accI[t] = vz; }

  for (int kc = 0; kc < K; kc += 32) {
    v16bf ar = load_a(Ar + (size_t)m0 * K + kc, K, lane);
    v16bf ai = load_a(Ai + (size_t)m0 * K + kc, K, lane);
#pragma unroll
    for (int t = 0; t < 4; ++t) {
      v16bf br = load_b(Wr + (size_t)(n0 + t * 16) * K + kc, K, lane);
      v16bf bi = load_b(Wi + (size_t)(n0 + t * 16) * K + kc, K, lane);
      accA[t] = WMMA_BF16(ar, br, accA[t]);   // Xr Wr
      accB[t] = WMMA_BF16(ai, bi, accB[t]);   // Xi Wi
      accI[t] = WMMA_BF16(ar, bi, accI[t]);   // Xr Wi
      accI[t] = WMMA_BF16(ai, br, accI[t]);   // + Xi Wr
    }
  }

  const int cb = lane & 15;
  const int rsh = (lane < 16) ? 0 : 8;
#pragma unroll
  for (int t = 0; t < 4; ++t) {
#pragma unroll
    for (int v = 0; v < 8; ++v) {
      float yr = accA[t][v] - accB[t][v];
      float yi = accI[t][v];
      int row = m0 + v + rsh;
      int col = n0 + t * 16 + cb;
      if (mode == 0) {
        int b = row >> 10, s = row & (S_ - 1);
        int h = col >> 7,  d = col & (D_ - 1);
        size_t idx = ((size_t)(b * H_ + h) * S_ + s) * D_ + d;
        outr[idx] = f2bf(yr); outi[idx] = f2bf(yi);
      } else if (mode == 1) {
        int b = row >> 10, s = row & (S_ - 1);
        int h = col >> 7,  d = col & (D_ - 1);
        size_t idx = ((size_t)(b * H_ + h) * D_ + d) * S_ + s;
        outr[idx] = f2bf(yr); outi[idx] = f2bf(yi);
      } else {
        size_t idx = (size_t)row * D_ + col;
        outc[2 * idx]     = yr;
        outc[2 * idx + 1] = yi;
      }
    }
  }
}

// ---------- complex-magnitude flash attention ----------
// Q,K: [B,H,S,D] bf16; VT: [B,H,D,S] bf16; O: [B,S,H*D] bf16.
// attn = softmax( scale * sqrt((QrKr^T - QiKi^T)^2 + (QrKi^T + QiKr^T)^2) )
// O = attn @ V (r and i separately).

__global__ __launch_bounds__(128) void cv_attn_kernel(
    const __bf16* __restrict__ Qr, const __bf16* __restrict__ Qi,
    const __bf16* __restrict__ Kr, const __bf16* __restrict__ Ki,
    const __bf16* __restrict__ VTr, const __bf16* __restrict__ VTi,
    __bf16* __restrict__ Or, __bf16* __restrict__ Oi)
{
  __shared__ __bf16 plds[4][16 * 64];   // per-wave P tile (C-layout -> A-layout transpose)

  const int lane = threadIdx.x & 31;
  const int wave = threadIdx.x >> 5;
  const int q0 = blockIdx.x * 64 + wave * 16;
  const int h  = blockIdx.y;
  const int b  = blockIdx.z;
  const int bh = b * H_ + h;
  const float scale = 0.0883883476483184f;   // 1/sqrt(128)

  // Preload Q fragments for this wave's 16 rows (all K chunks), plus -Qi.
  const __bf16* qrb = Qr + ((size_t)bh * S_ + q0) * D_;
  const __bf16* qib = Qi + ((size_t)bh * S_ + q0) * D_;
  v16bf qar[4], qai[4], qain[4];
#pragma unroll
  for (int kc = 0; kc < 4; ++kc) {
    qar[kc]  = load_a(qrb + kc * 32, D_, lane);
    qai[kc]  = load_a(qib + kc * 32, D_, lane);
    qain[kc] = negbf(qai[kc]);
  }

  v8f accr[8], acci[8];
  const v8f vz = {};
  float mrun[8], lrun[8];
#pragma unroll
  for (int dt = 0; dt < 8; ++dt) { accr[dt] = vz; acci[dt] = vz; }
#pragma unroll
  for (int v = 0; v < 8; ++v) { mrun[v] = -1e30f; lrun[v] = 0.f; }

  __bf16* pw = &plds[wave][0];

#pragma unroll 1
  for (int kt = 0; kt < S_; kt += 64) {
    // prefetch next key tile (one row per lane, both halves)
    if (kt + 64 < S_) {
      __builtin_prefetch(Kr + ((size_t)bh * S_ + kt + 64 + lane) * D_, 0, 0);
      __builtin_prefetch(Kr + ((size_t)bh * S_ + kt + 96 + lane) * D_, 0, 0);
      __builtin_prefetch(Ki + ((size_t)bh * S_ + kt + 64 + lane) * D_, 0, 0);
      __builtin_prefetch(Ki + ((size_t)bh * S_ + kt + 96 + lane) * D_, 0, 0);
    }

    // ---- complex scores: Sr = QrKr^T - QiKi^T, Si = QrKi^T + QiKr^T ----
    v8f sr[4], si[4];
#pragma unroll
    for (int ct = 0; ct < 4; ++ct) { sr[ct] = vz; si[ct] = vz; }
#pragma unroll
    for (int ct = 0; ct < 4; ++ct) {
      const __bf16* krt = Kr + ((size_t)bh * S_ + kt + ct * 16) * D_;
      const __bf16* kit = Ki + ((size_t)bh * S_ + kt + ct * 16) * D_;
#pragma unroll
      for (int kc = 0; kc < 4; ++kc) {
        v16bf bkr = load_b(krt + kc * 32, D_, lane);
        v16bf bki = load_b(kit + kc * 32, D_, lane);
        sr[ct] = WMMA_BF16(qar[kc],  bkr, sr[ct]);
        sr[ct] = WMMA_BF16(qain[kc], bki, sr[ct]);
        si[ct] = WMMA_BF16(qar[kc],  bki, si[ct]);
        si[ct] = WMMA_BF16(qai[kc],  bkr, si[ct]);
      }
    }

    // ---- magnitude + online softmax (row stats align with C-layout halves) ----
    float p[4][8];
    float mx[8];
#pragma unroll
    for (int v = 0; v < 8; ++v) {
      float m = -1e30f;
#pragma unroll
      for (int ct = 0; ct < 4; ++ct) {
        float mag = scale * sqrtf(sr[ct][v] * sr[ct][v] + si[ct][v] * si[ct][v]);
        p[ct][v] = mag;
        m = fmaxf(m, mag);
      }
#pragma unroll
      for (int off = 1; off < 16; off <<= 1)
        m = fmaxf(m, __shfl_xor(m, off, 32));
      mx[v] = m;
    }
#pragma unroll
    for (int v = 0; v < 8; ++v) {
      float mnew = fmaxf(mrun[v], mx[v]);
      float corr = __expf(mrun[v] - mnew);
      mrun[v] = mnew;
      float rs = 0.f;
#pragma unroll
      for (int ct = 0; ct < 4; ++ct) {
        float e = __expf(p[ct][v] - mnew);
        p[ct][v] = e;
        rs += e;
      }
#pragma unroll
      for (int off = 1; off < 16; off <<= 1)
        rs += __shfl_xor(rs, off, 32);
      lrun[v] = lrun[v] * corr + rs;
#pragma unroll
      for (int dt = 0; dt < 8; ++dt) { accr[dt][v] *= corr; acci[dt][v] *= corr; }
    }

    // ---- transpose P (C-layout regs -> A-layout via LDS) ----
    const int rbase = (lane < 16) ? 0 : 8;
    const int cb = lane & 15;
#pragma unroll
    for (int ct = 0; ct < 4; ++ct)
#pragma unroll
      for (int v = 0; v < 8; ++v)
        pw[(v + rbase) * 64 + ct * 16 + cb] = f2bf(p[ct][v]);
    // same-wave LDS ops are in-order; loads below see the stores above
    v16bf pa0 = load_a(pw, 64, lane);
    v16bf pa1 = load_a(pw + 32, 64, lane);

    // ---- accumulate P @ Vr, P @ Vi (VT rows are contiguous in keys) ----
#pragma unroll
    for (int dt = 0; dt < 8; ++dt) {
      const __bf16* vtr = VTr + ((size_t)bh * D_ + dt * 16) * S_ + kt;
      const __bf16* vti = VTi + ((size_t)bh * D_ + dt * 16) * S_ + kt;
      v16bf br0 = load_b(vtr,      S_, lane);
      v16bf br1 = load_b(vtr + 32, S_, lane);
      v16bf bi0 = load_b(vti,      S_, lane);
      v16bf bi1 = load_b(vti + 32, S_, lane);
      accr[dt] = WMMA_BF16(pa0, br0, accr[dt]);
      accr[dt] = WMMA_BF16(pa1, br1, accr[dt]);
      acci[dt] = WMMA_BF16(pa0, bi0, acci[dt]);
      acci[dt] = WMMA_BF16(pa1, bi1, acci[dt]);
    }
  }

  // ---- normalize and write O in [B,S,H*D] ----
  const int rbase = (lane < 16) ? 0 : 8;
  const int cb = lane & 15;
#pragma unroll
  for (int v = 0; v < 8; ++v) {
    float inv = 1.f / lrun[v];
    int s = q0 + v + rbase;
#pragma unroll
    for (int dt = 0; dt < 8; ++dt) {
      int d = h * D_ + dt * 16 + cb;
      size_t idx = ((size_t)b * S_ + s) * HD + d;
      Or[idx] = f2bf(accr[dt][v] * inv);
      Oi[idx] = f2bf(acci[dt][v] * inv);
    }
  }
}

// ---------- host launch ----------

extern "C" void kernel_launch(void* const* d_in, const int* in_sizes, int n_in,
                              void* d_out, int out_size, void* d_ws, size_t ws_size,
                              hipStream_t stream) {
  (void)in_sizes; (void)n_in; (void)out_size; (void)ws_size;

  __bf16* ws = (__bf16*)d_ws;
  size_t off = 0;
  auto alloc = [&](size_t n) { __bf16* p = ws + off; off += n; return p; };

  const size_t NX = (size_t)MROWS * D_;        // 524288  per input tensor
  const size_t NW = (size_t)HD * D_;           // 131072  per weight matrix
  const size_t NT = (size_t)B_ * H_ * S_ * D_; // 4194304 per head-major tensor

  __bf16* xb[6]; for (int i = 0; i < 6; ++i) xb[i] = alloc(NX);
  __bf16* wb[8]; for (int i = 0; i < 8; ++i) wb[i] = alloc(NW);
  __bf16 *qrB = alloc(NT), *qiB = alloc(NT);
  __bf16 *krB = alloc(NT), *kiB = alloc(NT);
  __bf16 *vtr = alloc(NT), *vti = alloc(NT);
  __bf16 *orB = alloc(NT), *oiB = alloc(NT);

  for (int i = 0; i < 6; ++i)
    cvt_kernel<<<dim3((unsigned)((NX + 255) / 256)), dim3(256), 0, stream>>>(
        (const float*)d_in[i], xb[i], (int)NX);
  for (int i = 0; i < 8; ++i)
    cvt_kernel<<<dim3((unsigned)((NW + 255) / 256)), dim3(256), 0, stream>>>(
        (const float*)d_in[6 + i], wb[i], (int)NW);

  dim3 blk(128);
  // Q, K projections -> [B,H,S,D]
  cgemm_kernel<<<dim3(MROWS / 64, HD / 64), blk, 0, stream>>>(
      xb[0], xb[1], wb[0], wb[1], D_, 0, qrB, qiB, nullptr);
  cgemm_kernel<<<dim3(MROWS / 64, HD / 64), blk, 0, stream>>>(
      xb[2], xb[3], wb[2], wb[3], D_, 0, krB, kiB, nullptr);
  // V projection -> transposed [B,H,D,S]
  cgemm_kernel<<<dim3(MROWS / 64, HD / 64), blk, 0, stream>>>(
      xb[4], xb[5], wb[4], wb[5], D_, 1, vtr, vti, nullptr);

  // attention
  cv_attn_kernel<<<dim3(S_ / 64, H_, B_), blk, 0, stream>>>(
      qrB, qiB, krB, kiB, vtr, vti, orB, oiB);

  // output projection -> interleaved complex64
  cgemm_kernel<<<dim3(MROWS / 64, D_ / 64), blk, 0, stream>>>(
      orB, oiB, wb[6], wb[7], HD, 2, nullptr, nullptr, (float*)d_out);
}
